// IoUPolyLoss_22471268893277
// MI455X (gfx1250) — compile-verified, matrix operand order checked by reading, not run on hardware
//
#include <hip/hip_runtime.h>

typedef __attribute__((ext_vector_type(16))) _Float16 v16h;
typedef __attribute__((ext_vector_type(8)))  float    v8f;

// Problem constants (fixed by setup_inputs in the reference)
constexpr int Bn  = 4;
constexpr int Cn  = 32;     // channels -> V = 16 vertices (x,y pairs)
constexpr int Hn  = 128;
constexpr int Wn  = 128;
constexpr int Kn  = 128;
constexpr int HW  = Hn * Wn;
constexpr int NV  = Cn / 2; // 16 vertices
constexpr int NPOLY = Bn * Kn; // 512

// Even-odd scanline parity for one row, bit-packed: bit px of m[] is
// "pixel (px,py) inside polygon".  XOR of prefix masks of length ceil(xint)
// over crossing edges == parity of #(px < xint).
__device__ __forceinline__ void raster_row(const float* __restrict__ vx,
                                           const float* __restrict__ vy,
                                           float py, unsigned m[4]) {
  m[0] = m[1] = m[2] = m[3] = 0u;
#pragma unroll
  for (int v = 0; v < NV; ++v) {
    const int vn = (v + 1) & (NV - 1);
    const float y1 = vy[v], y2 = vy[vn];
    const bool c1 = (y1 <= py);
    const bool c2 = (y2 <= py);
    if (c1 != c2) {                       // crossing edge (y1 != y2 guaranteed)
      const float x1 = vx[v], x2 = vx[vn];
      const float t    = (py - y1) / (y2 - y1);
      const float xint = x1 + t * (x2 - x1);
      int n = (int)ceilf(xint);           // #{integer px >= 0 : px < xint}
      n = n < 0 ? 0 : (n > Wn ? Wn : n);
#pragma unroll
      for (int w = 0; w < 4; ++w) {       // XOR prefix mask of length n
        const int r = n - (w << 5);
        const unsigned word =
            (r <= 0) ? 0u : ((r >= 32) ? 0xFFFFFFFFu : ((1u << r) - 1u));
        m[w] ^= word;
      }
    }
  }
}

// One block per polygon: gather 32 channels, build vertices, rasterize both
// polygons one row per thread with bit-packed parity, popcount + wave32
// shuffle reduction.
__global__ __launch_bounds__(Hn) void iou_raster_kernel(
    const float* __restrict__ output,   // [B,C,H,W]
    const int*   __restrict__ ind,      // [B,K]
    const float* __restrict__ target,   // [B,C,K]
    float*       __restrict__ iou_out)  // [B*K]
{
  __shared__ float pvx[NV], pvy[NV], gvx[NV], gvy[NV];
  __shared__ int   wcp[4], wcg[4], wci[4];   // one partial per wave32

  const int poly = blockIdx.x;
  const int b = poly >> 7;      // / Kn
  const int k = poly & (Kn - 1);
  const int t = threadIdx.x;

  if (t < Cn) {
    const int c   = t;
    const int idx = ind[b * Kn + k];
    const float pv = output[(b * Cn + c) * HW + idx];
    const float gv = target[(b * Cn + c) * Kn + k];
    const float fp = truncf(pv) + 100.0f;   // OFFSET
    const float fg = truncf(gv) + 100.0f;
    const int vtx = c >> 1;
    if (c & 1) { pvy[vtx] = fp; gvy[vtx] = fg; }
    else       { pvx[vtx] = fp; gvx[vtx] = fg; }
  }
  __syncthreads();

  const float py = (float)t;   // one row per thread
  unsigned pm[4], gm[4];
  raster_row(pvx, pvy, py, pm);
  raster_row(gvx, gvy, py, gm);

  int cp = 0, cg = 0, ci = 0;
#pragma unroll
  for (int w = 0; w < 4; ++w) {
    cp += __popc(pm[w]);
    cg += __popc(gm[w]);
    ci += __popc(pm[w] & gm[w]);
  }

  // wave32 reduction (warpSize == 32 on gfx1250)
#pragma unroll
  for (int off = 16; off > 0; off >>= 1) {
    cp += __shfl_down(cp, off);
    cg += __shfl_down(cg, off);
    ci += __shfl_down(ci, off);
  }
  const int wave = t >> 5;
  if ((t & 31) == 0) { wcp[wave] = cp; wcg[wave] = cg; wci[wave] = ci; }
  __syncthreads();

  if (t == 0) {
    const int scp = wcp[0] + wcp[1] + wcp[2] + wcp[3];
    const int scg = wcg[0] + wcg[1] + wcg[2] + wcg[3];
    const int sci = wci[0] + wci[1] + wci[2] + wci[3];
    const float inter = (float)sci;
    const float uni   = (float)(scp + scg) - inter;
    iou_out[poly] = inter / (uni + 1e-4f);
  }
}

// Single-wave exact reduction via WMMA:
//   w_j = iou_j * mask_j  split into f16 hi/lo (hi+lo reproduces f32 value).
//   A row0 = hi chunk, row1 = lo chunk, row2 = mask chunk, rows 3..15 = 0;
//   B column 0 = ones.  After 16 accumulating K=32 WMMAs:
//   D[0][0]+D[1][0] = sum(iou*mask), D[2][0] = sum(mask)  (f32 accumulate).
// B is constant along K, so any intra-lane K-ordering of the A operand is a
// bijection onto the chunk and the row sums are permutation-invariant.
// A operands are loaded as contiguous v16h vectors (branchless, b128 LDS).
__global__ __launch_bounds__(32) void iou_reduce_wmma(
    const float* __restrict__ iou,   // [NPOLY]
    const int*   __restrict__ mask,  // [NPOLY]
    float*       __restrict__ out)   // [1]
{
  // rows: 0 = hi, 1 = lo, 2 = mask, 3 = zeros (for A rows 3..15)
  __shared__ __align__(32) _Float16 s_dat[4 * NPOLY];
  const int t = threadIdx.x;

  for (int j = t; j < NPOLY; j += 32) {
    const float mf = (float)mask[j];
    const float w  = iou[j] * mf;          // exact: mf in {0,1}
    const _Float16 hi = (_Float16)w;
    const float    lo = w - (float)hi;
    s_dat[0 * NPOLY + j] = hi;
    s_dat[1 * NPOLY + j] = (_Float16)lo;
    s_dat[2 * NPOLY + j] = (_Float16)mf;
    s_dat[3 * NPOLY + j] = (_Float16)0.0f;
  }
  __syncthreads();

  const int mrow = t & 15;                 // lane -> M (A) / N (B)
  const int g    = t >> 4;                 // lane half -> K half of chunk
  const int rsel = (mrow <= 2) ? mrow : 3; // row select (3 = zero row)
  const _Float16* base = &s_dat[rsel * NPOLY + g * 16];

  v16h bv;                                 // B: column 0 = 1.0 for every K
#pragma unroll
  for (int s = 0; s < 16; ++s)
    bv[s] = (mrow == 0) ? (_Float16)1.0f : (_Float16)0.0f;

  v8f acc = {};
#pragma unroll
  for (int c = 0; c < NPOLY / 32; ++c) {   // 16 chained K=32 WMMAs
    const v16h av = *(const v16h*)(base + c * 32);  // 32B-aligned vector load
    acc = __builtin_amdgcn_wmma_f32_16x16x32_f16(
        /*neg_a=*/false, av, /*neg_b=*/false, bv,
        /*c_mod=*/(short)0, acc, /*reuse_a=*/false, /*reuse_b=*/false);
  }

  if (t == 0) {
    const float S = acc[0] + acc[1];   // D[0][0] + D[1][0] = sum(iou*mask)
    const float M = acc[2];            // D[2][0]           = sum(mask)
    out[0] = 1.0f - S / (M + 1e-4f);
  }
}

extern "C" void kernel_launch(void* const* d_in, const int* in_sizes, int n_in,
                              void* d_out, int out_size, void* d_ws, size_t ws_size,
                              hipStream_t stream) {
  const float* output = (const float*)d_in[0];  // [B,C,H,W] f32
  const int*   mask   = (const int*)  d_in[1];  // [B,K] i32
  const int*   ind    = (const int*)  d_in[2];  // [B,K] i32
  const float* target = (const float*)d_in[3];  // [B,C,K] f32
  float* iou_ws = (float*)d_ws;                 // NPOLY floats scratch
  float* loss   = (float*)d_out;

  iou_raster_kernel<<<dim3(NPOLY), dim3(Hn), 0, stream>>>(output, ind, target, iou_ws);
  iou_reduce_wmma  <<<dim3(1),     dim3(32), 0, stream>>>(iou_ws, mask, loss);
}
